// LocalSparseAttention_86268713108280
// MI455X (gfx1250) — matrix operand order repaired and checked by reading, not compiled, not run on hardware
//
#include <hip/hip_runtime.h>

#define DMODEL 512
#define LSEQ   2048
#define BATCH  2
#define NHEADS 8
#define HD     64
#define BHN    (BATCH * NHEADS)

typedef __attribute__((ext_vector_type(16))) _Float16 v16h;
typedef __attribute__((ext_vector_type(8)))  _Float16 v8h;
typedef __attribute__((ext_vector_type(8)))  float    v8f;

static __device__ __forceinline__ v8f wmma_f16(v16h a, v16h b, v8f c) {
  // D = A(16x32 f16) * B(32x16 f16) + C(16x16 f32)
  return __builtin_amdgcn_wmma_f32_16x16x32_f16(false, a, false, b, (short)0, c,
                                                false, false);
}

// A-fragment (16x32, f16). Rows contiguous-with-stride.
// Lane l (<16): row=l, K = {0..7, 16..23}; lane l (>=16): row=l-16, K = {8..15, 24..31}.
static __device__ __forceinline__ v16h load_frag_a(const _Float16* base, int stride) {
  const int lane = threadIdx.x & 31;
  const int r    = lane & 15;
  const int hi   = lane >> 4;
  const _Float16* p = base + (size_t)r * stride + hi * 8;
  v8h lo = *(const v8h*)(p);        // K = hi*8 + 0..7
  v8h hh = *(const v8h*)(p + 16);   // K = 16 + hi*8 + 0..7
  v16h out;
#pragma unroll
  for (int i = 0; i < 8; ++i) { out[i] = lo[i]; out[8 + i] = hh[i]; }
  return out;
}

// B-fragment (32x16, f16). Column n of B contiguous at base + n*stride + k.
// Lane l: col = l&15, K = (l>>4)*8 + {0..7, 16..23}.
static __device__ __forceinline__ v16h load_frag_b(const _Float16* base, int stride) {
  const int lane = threadIdx.x & 31;
  const int n    = lane & 15;
  const int hi   = lane >> 4;
  const _Float16* p = base + (size_t)n * stride + hi * 8;
  v8h lo = *(const v8h*)(p);
  v8h hh = *(const v8h*)(p + 16);
  v16h out;
#pragma unroll
  for (int i = 0; i < 8; ++i) { out[i] = lo[i]; out[8 + i] = hh[i]; }
  return out;
}

// ---------------------------------------------------------------- convert
__global__ void k_cvt(const float* __restrict__ in, _Float16* __restrict__ out, int n) {
  int i = blockIdx.x * blockDim.x + threadIdx.x;
  if (i < n) out[i] = (_Float16)in[i];
}

// ---------------------------------------------------------------- QKV GEMM
// y[m, n] = sum_k xh[m,k] * W[n,k] (+ bias), m in [0,4096), n in [0,1536)
// Wave tile: 32 rows x 64 cols; ping-pong double-buffered over K (no reg copies).
// scatter: q -> Qh[bh][l][d] (scaled 1/8), k -> Kh[bh][l][d], v -> Vt[bh][d][l]
__global__ __launch_bounds__(256) void k_qkv(const _Float16* __restrict__ xh,
                                             const _Float16* __restrict__ wh,
                                             const float* __restrict__ bias,
                                             _Float16* __restrict__ Qh,
                                             _Float16* __restrict__ Kh,
                                             _Float16* __restrict__ Vt) {
  const int lane = threadIdx.x & 31;
  const int wave = threadIdx.x >> 5;
  const int wg   = blockIdx.x * 8 + wave;
  const int NT   = (3 * DMODEL) / 64;  // 24 n-tiles of 64
  const int mt   = wg / NT;            // 0..127 (tiles of 32 rows)
  const int nt   = wg % NT;
  const int mbase = mt * 32;
  const int nbase = nt * 64;

  const _Float16* ap0 = xh + (size_t)mbase * DMODEL;
  const _Float16* ap1 = xh + (size_t)(mbase + 16) * DMODEL;
  const _Float16* bp  = wh + (size_t)nbase * DMODEL;

  v8f c[2][4] = {};
  v16h a0[2], a1[2], b[2][4];

  auto load_stage = [&](int s, int kt) {
    a0[s] = load_frag_a(ap0 + kt, DMODEL);
    a1[s] = load_frag_a(ap1 + kt, DMODEL);
#pragma unroll
    for (int j = 0; j < 4; ++j)
      b[s][j] = load_frag_b(bp + (size_t)(j * 16) * DMODEL + kt, DMODEL);
  };
  auto compute_stage = [&](int s) {
#pragma unroll
    for (int j = 0; j < 4; ++j) c[0][j] = wmma_f16(a0[s], b[s][j], c[0][j]);
#pragma unroll
    for (int j = 0; j < 4; ++j) c[1][j] = wmma_f16(a1[s], b[s][j], c[1][j]);
  };

  load_stage(0, 0);
  for (int kt = 0; kt < DMODEL - 64; kt += 64) {
    load_stage(1, kt + 32);
    compute_stage(0);
    load_stage(0, kt + 64);
    compute_stage(1);
  }
  load_stage(1, DMODEL - 32);
  compute_stage(0);
  compute_stage(1);

  // epilogue: bias + scatter into attention-friendly layouts
  const int r0 = lane & 15;
  const int hi = lane >> 4;
#pragma unroll
  for (int j = 0; j < 4; ++j) {
    const int n   = nbase + j * 16 + r0;
    const int seg = n / DMODEL;   // 0=q 1=k 2=v
    const int dd  = n % DMODEL;
    const int h   = dd >> 6;
    const int d   = dd & (HD - 1);
    const float bn = bias[n];
#pragma unroll
    for (int rg = 0; rg < 2; ++rg) {
#pragma unroll
      for (int r = 0; r < 8; ++r) {
        const int m    = mbase + rg * 16 + r + hi * 8;
        const int bidx = m >> 11;           // m / LSEQ
        const int l    = m & (LSEQ - 1);
        const int bh   = bidx * NHEADS + h;
        const float val = c[rg][j][r] + bn;
        if (seg == 0)
          Qh[((size_t)bh * LSEQ + l) * HD + d] = (_Float16)(val * 0.125f);  // 1/sqrt(64)
        else if (seg == 1)
          Kh[((size_t)bh * LSEQ + l) * HD + d] = (_Float16)val;
        else
          Vt[((size_t)bh * HD + d) * LSEQ + l] = (_Float16)val;
      }
    }
  }
}

// ---------------------------------------------------------------- flash attention
// grid: (LSEQ/64, BHN), block: 128 (4 waves, each owning 16 query rows).
// Key-tile loop unrolled by 2 with ping-pong K-fragment buffers; V fragments are
// issued right after the S WMMAs so they fly during the softmax VALU block.
__global__ __launch_bounds__(128) void k_attn(const _Float16* __restrict__ Qh,
                                              const _Float16* __restrict__ Kh,
                                              const _Float16* __restrict__ Vt,
                                              _Float16* __restrict__ ctxh,
                                              const int* __restrict__ winp) {
  __shared__ __align__(16) _Float16 lds[4][16 * 32];
  const int lane  = threadIdx.x & 31;
  const int wave  = threadIdx.x >> 5;
  const int bh    = blockIdx.y;
  const int qbase = blockIdx.x * 64 + wave * 16;
  const int half  = winp[0] >> 1;

  const _Float16* Qp = Qh + (size_t)bh * LSEQ * HD;
  const _Float16* Kp = Kh + (size_t)bh * LSEQ * HD;
  const _Float16* Vp = Vt + (size_t)bh * HD * LSEQ;

  const v16h qa0 = load_frag_a(Qp + (size_t)qbase * HD + 0,  HD);
  const v16h qa1 = load_frag_a(Qp + (size_t)qbase * HD + 32, HD);

  const int r0 = lane & 15;
  const int hi = lane >> 4;
  _Float16* myLds = &lds[wave][0];

  float mrow[8], lrow[8];
  v8f acc[4] = {};
#pragma unroll
  for (int r = 0; r < 8; ++r) { mrow[r] = -__builtin_inff(); lrow[r] = 0.0f; }

  v16h kbuf[2][4];
  auto loadK = [&](int s, int kb) {
    const _Float16* kp = Kp + (size_t)kb * HD;
    kbuf[s][0] = load_frag_b(kp + 0,            HD);
    kbuf[s][1] = load_frag_b(kp + 32,           HD);
    kbuf[s][2] = load_frag_b(kp + 16 * HD,      HD);
    kbuf[s][3] = load_frag_b(kp + 16 * HD + 32, HD);
  };

  auto process = [&](int kb, int s) {
    // S = Q Kt for 32 keys (two 16-col halves)
    v8f s0 = {}, s1 = {};
    s0 = wmma_f16(qa0, kbuf[s][0], s0);
    s0 = wmma_f16(qa1, kbuf[s][1], s0);
    s1 = wmma_f16(qa0, kbuf[s][2], s1);
    s1 = wmma_f16(qa1, kbuf[s][3], s1);

    // V fragments for this tile: issue now, consumed after softmax
    v16h vf[4];
#pragma unroll
    for (int jj = 0; jj < 4; ++jj)
      vf[jj] = load_frag_b(Vp + (size_t)(jj * 16) * LSEQ + kb, LSEQ);

    // mask: ban j in [i-half, i+half)
    const int j0 = kb + r0;
    const int j1 = kb + 16 + r0;
#pragma unroll
    for (int r = 0; r < 8; ++r) {
      const int i = qbase + r + hi * 8;
      if (j0 >= i - half && j0 < i + half) s0[r] = -__builtin_inff();
      if (j1 >= i - half && j1 < i + half) s1[r] = -__builtin_inff();
    }

    // online softmax (row reductions across the 16-lane half-group)
    float p0[8], p1[8];
#pragma unroll
    for (int r = 0; r < 8; ++r) {
      float mx = fmaxf(s0[r], s1[r]);
#pragma unroll
      for (int off = 1; off < 16; off <<= 1) mx = fmaxf(mx, __shfl_xor(mx, off, 32));
      const float mnew  = fmaxf(mrow[r], mx);
      const float scale = __expf(mrow[r] - mnew);
      p0[r] = __expf(s0[r] - mnew);
      p1[r] = __expf(s1[r] - mnew);
      float rs = p0[r] + p1[r];
#pragma unroll
      for (int off = 1; off < 16; off <<= 1) rs += __shfl_xor(rs, off, 32);
      lrow[r] = lrow[r] * scale + rs;
      mrow[r] = mnew;
#pragma unroll
      for (int jj = 0; jj < 4; ++jj) acc[jj][r] = acc[jj][r] * scale;
    }

    // C-layout -> A-layout transpose of P through LDS
#pragma unroll
    for (int r = 0; r < 8; ++r) {
      const int row = r + hi * 8;
      myLds[row * 32 + r0]      = (_Float16)p0[r];
      myLds[row * 32 + 16 + r0] = (_Float16)p1[r];
    }
    __syncthreads();
    const v16h pa = load_frag_a(myLds, 32);
    __syncthreads();

    // O += P (16x32) @ V (32x64)
#pragma unroll
    for (int jj = 0; jj < 4; ++jj) acc[jj] = wmma_f16(pa, vf[jj], acc[jj]);
  };

  loadK(0, 0);
  for (int kb = 0; kb < LSEQ; kb += 64) {
    loadK(1, kb + 32);                     // always < LSEQ
    process(kb, 0);
    loadK(0, (kb + 64) & (LSEQ - 1));      // wraps harmlessly on last iter
    process(kb + 32, 1);
  }

  // normalize + scatter to ctx [b][l][h*64+d] as f16 for the out-proj GEMM
  const int bidx = bh >> 3;
  const int h    = bh & (NHEADS - 1);
#pragma unroll
  for (int jj = 0; jj < 4; ++jj) {
#pragma unroll
    for (int r = 0; r < 8; ++r) {
      const int i = qbase + r + hi * 8;
      const float o = acc[jj][r] / lrow[r];
      ctxh[((size_t)bidx * LSEQ + i) * DMODEL + h * HD + jj * 16 + r0] = (_Float16)o;
    }
  }
}

// ---------------------------------------------------------------- out proj + residual
// Wave tile: 32 rows x 64 cols; ping-pong double-buffered over K.
__global__ __launch_bounds__(256) void k_oproj(const _Float16* __restrict__ ctxh,
                                               const _Float16* __restrict__ wh,
                                               const float* __restrict__ bias,
                                               const float* __restrict__ x,
                                               float* __restrict__ y) {
  const int lane = threadIdx.x & 31;
  const int wave = threadIdx.x >> 5;
  const int wg   = blockIdx.x * 8 + wave;
  const int NT   = DMODEL / 64;  // 8
  const int mt   = wg / NT;      // 0..127
  const int nt   = wg % NT;
  const int mbase = mt * 32, nbase = nt * 64;

  const _Float16* ap0 = ctxh + (size_t)mbase * DMODEL;
  const _Float16* ap1 = ctxh + (size_t)(mbase + 16) * DMODEL;
  const _Float16* bp  = wh + (size_t)nbase * DMODEL;

  v8f c[2][4] = {};
  v16h a0[2], a1[2], b[2][4];

  auto load_stage = [&](int s, int kt) {
    a0[s] = load_frag_a(ap0 + kt, DMODEL);
    a1[s] = load_frag_a(ap1 + kt, DMODEL);
#pragma unroll
    for (int j = 0; j < 4; ++j)
      b[s][j] = load_frag_b(bp + (size_t)(j * 16) * DMODEL + kt, DMODEL);
  };
  auto compute_stage = [&](int s) {
#pragma unroll
    for (int j = 0; j < 4; ++j) c[0][j] = wmma_f16(a0[s], b[s][j], c[0][j]);
#pragma unroll
    for (int j = 0; j < 4; ++j) c[1][j] = wmma_f16(a1[s], b[s][j], c[1][j]);
  };

  load_stage(0, 0);
  for (int kt = 0; kt < DMODEL - 64; kt += 64) {
    load_stage(1, kt + 32);
    compute_stage(0);
    load_stage(0, kt + 64);
    compute_stage(1);
  }
  load_stage(1, DMODEL - 32);
  compute_stage(0);
  compute_stage(1);

  const int r0 = lane & 15, hi = lane >> 4;
#pragma unroll
  for (int j = 0; j < 4; ++j) {
    const int n = nbase + j * 16 + r0;
    const float bn = bias[n];
#pragma unroll
    for (int rg = 0; rg < 2; ++rg) {
#pragma unroll
      for (int r = 0; r < 8; ++r) {
        const int m = mbase + rg * 16 + r + hi * 8;
        y[(size_t)m * DMODEL + n] = c[rg][j][r] + bn + x[(size_t)m * DMODEL + n];
      }
    }
  }
}

// ---------------------------------------------------------------- LayerNorm
__global__ __launch_bounds__(256) void k_ln(const float* __restrict__ y,
                                            const float* __restrict__ gamma,
                                            const float* __restrict__ beta,
                                            float* __restrict__ out) {
  __shared__ float sm[8];
  const int m = blockIdx.x;
  const int t = threadIdx.x;
  const float* row = y + (size_t)m * DMODEL;
  const float v0 = row[t];
  const float v1 = row[t + 256];

  float s = v0 + v1;
#pragma unroll
  for (int off = 16; off; off >>= 1) s += __shfl_xor(s, off, 32);
  if ((t & 31) == 0) sm[t >> 5] = s;
  __syncthreads();
  float mean = 0.0f;
#pragma unroll
  for (int i = 0; i < 8; ++i) mean += sm[i];
  mean *= (1.0f / DMODEL);
  __syncthreads();

  const float d0 = v0 - mean, d1 = v1 - mean;
  float q = d0 * d0 + d1 * d1;
#pragma unroll
  for (int off = 16; off; off >>= 1) q += __shfl_xor(q, off, 32);
  if ((t & 31) == 0) sm[t >> 5] = q;
  __syncthreads();
  float var = 0.0f;
#pragma unroll
  for (int i = 0; i < 8; ++i) var += sm[i];
  var *= (1.0f / DMODEL);
  const float inv = rsqrtf(var + 1e-5f);

  out[(size_t)m * DMODEL + t]       = d0 * inv * gamma[t] + beta[t];
  out[(size_t)m * DMODEL + t + 256] = d1 * inv * gamma[t + 256] + beta[t + 256];
}

// ---------------------------------------------------------------- launch
extern "C" void kernel_launch(void* const* d_in, const int* in_sizes, int n_in,
                              void* d_out, int out_size, void* d_ws, size_t ws_size,
                              hipStream_t stream) {
  (void)in_sizes; (void)n_in; (void)out_size; (void)ws_size;

  const float* x     = (const float*)d_in[0];
  const float* w_in  = (const float*)d_in[1];
  const float* b_in  = (const float*)d_in[2];
  const float* w_out = (const float*)d_in[3];
  const float* b_out = (const float*)d_in[4];
  const float* gamma = (const float*)d_in[5];
  const float* beta  = (const float*)d_in[6];
  const int*   wsz   = (const int*)d_in[7];

  char* ws = (char*)d_ws;
  size_t off = 0;
  auto alloc = [&](size_t bytes) -> void* {
    void* p = ws + off;
    off += (bytes + 255) & ~(size_t)255;
    return p;
  };
  _Float16* xh   = (_Float16*)alloc((size_t)BATCH * LSEQ * DMODEL * 2);
  _Float16* wih  = (_Float16*)alloc((size_t)3 * DMODEL * DMODEL * 2);
  _Float16* woh  = (_Float16*)alloc((size_t)DMODEL * DMODEL * 2);
  _Float16* Qh   = (_Float16*)alloc((size_t)BHN * LSEQ * HD * 2);
  _Float16* Kh   = (_Float16*)alloc((size_t)BHN * LSEQ * HD * 2);
  _Float16* Vt   = (_Float16*)alloc((size_t)BHN * HD * LSEQ * 2);
  _Float16* ctxh = (_Float16*)alloc((size_t)BATCH * LSEQ * DMODEL * 2);
  float*    yb   = (float*)alloc((size_t)BATCH * LSEQ * DMODEL * 4);

  const int n1 = BATCH * LSEQ * DMODEL;
  const int n2 = 3 * DMODEL * DMODEL;
  const int n3 = DMODEL * DMODEL;
  k_cvt<<<(n1 + 255) / 256, 256, 0, stream>>>(x, xh, n1);
  k_cvt<<<(n2 + 255) / 256, 256, 0, stream>>>(w_in, wih, n2);
  k_cvt<<<(n3 + 255) / 256, 256, 0, stream>>>(w_out, woh, n3);

  // 128 m-tiles (32 rows) * 24 n-tiles (64 cols) = 3072 waves / 8 per block
  k_qkv<<<384, 256, 0, stream>>>(xh, wih, b_in, Qh, Kh, Vt);

  dim3 g_attn(LSEQ / 64, BHN);
  k_attn<<<g_attn, 128, 0, stream>>>(Qh, Kh, Vt, ctxh, wsz);

  // 128 m-tiles * 8 n-tiles = 1024 waves / 8 per block
  k_oproj<<<128, 256, 0, stream>>>(ctxh, woh, b_out, x, yb);

  k_ln<<<BATCH * LSEQ, 256, 0, stream>>>(yb, gamma, beta, (float*)d_out);
}
